// MultiHeadAttention_78443282694529
// MI455X (gfx1250) — matrix operand order repaired and checked by reading, not compiled, not run on hardware
//
#include <hip/hip_runtime.h>

// MHA forward: B=4, S=2048, D=1024, H=16, DK=64 on gfx1250 (MI455X).
// f16 WMMA (v_wmma_f32_16x16x32_f16) everywhere, f32 accumulate.
// Pipeline: cvt(f32->f16) -> QKV GEMM -> flash attention (TDM-staged K/V) -> out GEMM.

#define B_SZ   4
#define S_LEN  2048
#define D_DIM  1024
#define NH     16
#define DKH    64

typedef __attribute__((ext_vector_type(16))) _Float16     v16h;
typedef __attribute__((ext_vector_type(8)))  float        v8f;
typedef __attribute__((ext_vector_type(4)))  unsigned int v4u;
typedef __attribute__((ext_vector_type(8)))  int          v8i;
typedef __attribute__((ext_vector_type(4)))  int          v4i;

#if __has_builtin(__builtin_amdgcn_tensor_load_to_lds)
#define HAVE_TDM 1
#else
#define HAVE_TDM 0
#endif

static __device__ __forceinline__ v8f wmma_f16(v16h a, v16h b, v8f c) {
  return __builtin_amdgcn_wmma_f32_16x16x32_f16(false, a, false, b, (short)0, c,
                                                false, false);
}

// ---- Fragment loaders, contiguous k = 16*half + s order, A and B agree (wave32) ----------------
// A (16x32 f16): lanes 0-15 row M=lane, a[0..15]=K(k0..k0+15);
// lanes 16-31: K(k0+16..k0+31).
static __device__ __forceinline__ v16h load_a_f16(const _Float16* X, int ld,
                                                  int row0, int k0, int lane) {
  int m  = row0 + (lane & 15);
  int hf = lane >> 4;
  const _Float16* p = X + (size_t)m * ld + k0 + hf * 16;
  v16h a;
#pragma unroll
  for (int j = 0; j < 8; ++j) a[j] = p[j];
#pragma unroll
  for (int j = 0; j < 8; ++j) a[8 + j] = p[8 + j];
  return a;
}

// B (32x16): B[k][n] = Wm[n][k]; lanes 0-15: N=lane, k=k0..k0+15;
// lanes 16-31: N=lane-16, k=k0+16..k0+31. 16 contiguous f16 -> 32B vector load.
static __device__ __forceinline__ v16h load_b_f16(const _Float16* Wm, int ld,
                                                  int n0, int k0, int lane) {
  const _Float16* p =
      Wm + (size_t)(n0 + (lane & 15)) * ld + k0 + (lane >> 4) * 16;
  return *(const v16h*)p;  // 32B-aligned at every call site
}

// Same B layout sourced from LDS (generic pointer into __shared__ -> ds_load).
static __device__ __forceinline__ v16h load_b_lds(const _Float16* base, int ld,
                                                  int n0, int k0, int lane) {
  const _Float16* p = base + (n0 + (lane & 15)) * ld + k0 + (lane >> 4) * 16;
  v16h b;
#pragma unroll
  for (int j = 0; j < 16; ++j) b[j] = p[j];
  return b;
}

// ---------------------------------------------------------------------------
// TDM: 2D tile load (f16 elements) global -> LDS, per ISA ch.8 D# layout.
// This toolchain exposes the 6-arg builtin:
//   (uint32x4 g0, int32x8 g1, int32x4, int32x4, int32x8, i32 cpol)
// ---------------------------------------------------------------------------
#if HAVE_TDM
static __device__ __forceinline__ void tdm_load_2d(unsigned lds_off,
                                                   const _Float16* gptr,
                                                   unsigned tile_d0,
                                                   unsigned tile_d1,
                                                   unsigned tens_d0,
                                                   unsigned tens_d1,
                                                   unsigned stride0) {
  unsigned long long ga = (unsigned long long)(size_t)gptr;
  v4u g0;
  g0.x = 1u;                                            // count=1, user mode
  g0.y = lds_off;                                       // lds_addr [63:32]
  g0.z = (unsigned)ga;                                  // global_addr[31:0]
  g0.w = (unsigned)((ga >> 32) & 0x01FFFFFFu) | (2u << 30);  // addr[56:32]|type=2
  v8i g1;
  g1[0] = (int)(1u << 16);                              // data_size=1 (2 bytes)
  g1[1] = (int)((tens_d0 & 0xFFFFu) << 16);             // tensor_dim0[15:0]
  g1[2] = (int)((tens_d0 >> 16) | ((tens_d1 & 0xFFFFu) << 16));
  g1[3] = (int)((tens_d1 >> 16) | (tile_d0 << 16));     // tile_dim0
  g1[4] = (int)(tile_d1 & 0xFFFFu);                     // tile_dim1 (tile_dim2=0)
  g1[5] = (int)stride0;                                 // tensor_dim0_stride[31:0]
  g1[6] = 0;
  g1[7] = 0;
  v4i gz4 = {0, 0, 0, 0};
  v8i gz8 = {0, 0, 0, 0, 0, 0, 0, 0};
  __builtin_amdgcn_tensor_load_to_lds(g0, g1, gz4, gz4, gz8, 0);
}
#else
// Fallback: cooperative VMEM->LDS copy of a tile whose rows are contiguous.
static __device__ __forceinline__ void stage_rows(_Float16* dst,
                                                  const _Float16* src, int rows,
                                                  int rowelems, int srcstride,
                                                  int lane) {
  int cpr = (rowelems * 2) / 16;  // 16B chunks per row
  int total = rows * cpr;
  for (int c = lane; c < total; c += 32) {
    int r = c / cpr, q = c % cpr;
    ((uint4*)(dst + (size_t)r * rowelems))[q] =
        ((const uint4*)(src + (size_t)r * srcstride))[q];
  }
}
#endif

// ---------------------------------------------------------------------------
// Kernel 0: f32 -> f16 convert (bandwidth pass; n multiple of 4)
// ---------------------------------------------------------------------------
__global__ void cvt_kernel(const float* __restrict__ src,
                           _Float16* __restrict__ dst, int n) {
  int i = (blockIdx.x * blockDim.x + threadIdx.x) * 4;
  if (i < n) {
    float4 v = *(const float4*)(src + i);
    dst[i + 0] = (_Float16)v.x;
    dst[i + 1] = (_Float16)v.y;
    dst[i + 2] = (_Float16)v.z;
    dst[i + 3] = (_Float16)v.w;
  }
}

// ---------------------------------------------------------------------------
// Kernel 1: qkv = x @ Wqkv^T + bqkv; scatter to Q(scaled)[B,H,S,DK],
// K[B,H,S,DK], V transposed [B,H,DK,S]. One wave -> 32x64 tile.
// ---------------------------------------------------------------------------
__global__ void qkv_kernel(const _Float16* __restrict__ xh,
                           const _Float16* __restrict__ Wh,
                           const float* __restrict__ bias,
                           _Float16* __restrict__ Qh, _Float16* __restrict__ Kh,
                           _Float16* __restrict__ Vt) {
  const int lane = threadIdx.x;
  const int m0 = blockIdx.x * 32;
  const int n0 = blockIdx.y * 64;
  v8f acc[2][4] = {};
  for (int k0 = 0; k0 < D_DIM; k0 += 32) {
    v16h a0 = load_a_f16(xh, D_DIM, m0, k0, lane);
    v16h a1 = load_a_f16(xh, D_DIM, m0 + 16, k0, lane);
#pragma unroll
    for (int t = 0; t < 4; ++t) {
      v16h b = load_b_f16(Wh, D_DIM, n0 + t * 16, k0, lane);
      acc[0][t] = wmma_f16(a0, b, acc[0][t]);
      acc[1][t] = wmma_f16(a1, b, acc[1][t]);
    }
  }
  const int rb = (lane >> 4) * 8, cl = lane & 15;
#pragma unroll
  for (int t = 0; t < 4; ++t) {
    int n = n0 + t * 16 + cl;
    float bv = bias[n];
    int which = n >> 10;  // 0=Q 1=K 2=V (uniform per block)
    int h = (n & 1023) >> 6;
    int dk = n & 63;
#pragma unroll
    for (int u = 0; u < 2; ++u)
#pragma unroll
      for (int r = 0; r < 8; ++r) {
        int m = m0 + u * 16 + rb + r;
        int b = m >> 11, s = m & (S_LEN - 1);
        float v = acc[u][t][r] + bv;
        size_t bh = (size_t)(b * NH + h);
        if (which == 0)
          Qh[(bh * S_LEN + s) * DKH + dk] = (_Float16)(v * 0.125f);
        else if (which == 1)
          Kh[(bh * S_LEN + s) * DKH + dk] = (_Float16)v;
        else
          Vt[(bh * DKH + dk) * S_LEN + s] = (_Float16)v;
      }
  }
}

// ---------------------------------------------------------------------------
// Kernel 2: flash attention, one wave per (b*H+h, 16-query tile).
// K/V tiles double-buffered into LDS via TDM; softmax overlapped with DMA.
// ---------------------------------------------------------------------------
__global__ void attn_kernel(const _Float16* __restrict__ Qh,
                            const _Float16* __restrict__ Kh,
                            const _Float16* __restrict__ Vt,
                            _Float16* __restrict__ attnW) {
  const int lane = threadIdx.x;
  const int bh = blockIdx.y;
  const int q0 = blockIdx.x * 16;
  const _Float16* Q = Qh + (size_t)bh * S_LEN * DKH;
  const _Float16* K = Kh + (size_t)bh * S_LEN * DKH;
  const _Float16* V = Vt + (size_t)bh * DKH * S_LEN;

  __shared__ _Float16 Kb[2][32 * DKH];  // 32 keys x 64 dk
  __shared__ _Float16 Vb[2][DKH * 32];  // 64 dk rows x 32 keys
  __shared__ float Pld[16][33];

  v16h qa0 = load_a_f16(Q, DKH, q0, 0, lane);
  v16h qa1 = load_a_f16(Q, DKH, q0, 32, lane);

  float mrun[8], lsum[8];
#pragma unroll
  for (int r = 0; r < 8; ++r) { mrun[r] = -3.0e38f; lsum[r] = 0.0f; }
  v8f o[4] = {};
  const int rb = (lane >> 4) * 8, cl = lane & 15;

#if HAVE_TDM
  // preload chunk 0 into buffer 0
  tdm_load_2d((unsigned)(size_t)&Kb[0][0], K, DKH, 32, DKH, S_LEN, DKH);
  tdm_load_2d((unsigned)(size_t)&Vb[0][0], V, 32, DKH, S_LEN, DKH, S_LEN);
#endif

  const int nchunks = S_LEN / 32;
  for (int ic = 0; ic < nchunks; ++ic) {
    const int buf = ic & 1;
#if HAVE_TDM
    if (ic + 1 < nchunks) {
      int kc2 = (ic + 1) * 32;
      tdm_load_2d((unsigned)(size_t)&Kb[buf ^ 1][0], K + (size_t)kc2 * DKH,
                  DKH, 32, DKH, S_LEN, DKH);
      tdm_load_2d((unsigned)(size_t)&Vb[buf ^ 1][0], V + kc2, 32, DKH, S_LEN,
                  DKH, S_LEN);
      __builtin_amdgcn_s_wait_tensorcnt(2);  // current buffer's 2 DMAs done
    } else {
      __builtin_amdgcn_s_wait_tensorcnt(0);
    }
    __asm__ __volatile__("" ::: "memory");
#else
    int kc = ic * 32;
    stage_rows(&Kb[buf][0], K + (size_t)kc * DKH, 32, DKH, DKH, lane);
    stage_rows(&Vb[buf][0], V + kc, DKH, 32, S_LEN, lane);
    __syncthreads();
#endif

    // scores (Q pre-scaled by 1/sqrt(DK)); B operands from LDS
    v8f s0 = {}, s1 = {};
    s0 = wmma_f16(qa0, load_b_lds(&Kb[buf][0], DKH, 0, 0, lane), s0);
    s0 = wmma_f16(qa1, load_b_lds(&Kb[buf][0], DKH, 0, 32, lane), s0);
    s1 = wmma_f16(qa0, load_b_lds(&Kb[buf][0], DKH, 16, 0, lane), s1);
    s1 = wmma_f16(qa1, load_b_lds(&Kb[buf][0], DKH, 16, 32, lane), s1);

    // online softmax; rows live per 16-lane half -> width-16 shuffles
#pragma unroll
    for (int r = 0; r < 8; ++r) {
      float mc = fmaxf(s0[r], s1[r]);
#pragma unroll
      for (int off = 1; off < 16; off <<= 1)
        mc = fmaxf(mc, __shfl_xor(mc, off, 16));
      float mn = fmaxf(mrun[r], mc);
      float corr = __expf(mrun[r] - mn);
      mrun[r] = mn;
      float p0 = __expf(s0[r] - mn);
      float p1 = __expf(s1[r] - mn);
      s0[r] = p0; s1[r] = p1;
      float ps = p0 + p1;
#pragma unroll
      for (int off = 1; off < 16; off <<= 1)
        ps += __shfl_xor(ps, off, 16);
      lsum[r] = lsum[r] * corr + ps;
      o[0][r] *= corr; o[1][r] *= corr; o[2][r] *= corr; o[3][r] *= corr;
    }

    // transpose P from C-layout to A-layout through LDS
#pragma unroll
    for (int r = 0; r < 8; ++r) {
      Pld[rb + r][cl] = s0[r];
      Pld[rb + r][16 + cl] = s1[r];
    }
    __syncthreads();
    v16h pa;
    {
      int m = lane & 15, hf = lane >> 4;
#pragma unroll
      for (int j = 0; j < 8; ++j) pa[j] = (_Float16)Pld[m][hf * 16 + j];
#pragma unroll
      for (int j = 0; j < 8; ++j) pa[8 + j] = (_Float16)Pld[m][hf * 16 + 8 + j];
    }
    __syncthreads();

    // O += P @ V (Vb rows = dk, contiguous 32 keys)
#pragma unroll
    for (int t = 0; t < 4; ++t)
      o[t] = wmma_f16(pa, load_b_lds(&Vb[buf][0], 32, t * 16, 0, lane), o[t]);
  }

  // finalize, write merged-head f16 [B,S,D]
  const int b = bh >> 4, h = bh & 15;
#pragma unroll
  for (int t = 0; t < 4; ++t)
#pragma unroll
    for (int r = 0; r < 8; ++r) {
      int srow = q0 + rb + r;
      int col = h * DKH + t * 16 + cl;
      float val = o[t][r] / lsum[r];
      attnW[((size_t)b * S_LEN + srow) * D_DIM + col] = (_Float16)val;
    }
}

// ---------------------------------------------------------------------------
// Kernel 3: out = attn @ Wout^T + bout (fp32 out). One wave -> 32x64 tile.
// ---------------------------------------------------------------------------
__global__ void proj_kernel(const _Float16* __restrict__ A,
                            const _Float16* __restrict__ Wh,
                            const float* __restrict__ bias,
                            float* __restrict__ out) {
  const int lane = threadIdx.x;
  const int m0 = blockIdx.x * 32;
  const int n0 = blockIdx.y * 64;
  v8f acc[2][4] = {};
  for (int k0 = 0; k0 < D_DIM; k0 += 32) {
    v16h a0 = load_a_f16(A, D_DIM, m0, k0, lane);
    v16h a1 = load_a_f16(A, D_DIM, m0 + 16, k0, lane);
#pragma unroll
    for (int t = 0; t < 4; ++t) {
      v16h b = load_b_f16(Wh, D_DIM, n0 + t * 16, k0, lane);
      acc[0][t] = wmma_f16(a0, b, acc[0][t]);
      acc[1][t] = wmma_f16(a1, b, acc[1][t]);
    }
  }
  const int rb = (lane >> 4) * 8, cl = lane & 15;
#pragma unroll
  for (int t = 0; t < 4; ++t) {
    int n = n0 + t * 16 + cl;
    float bv = bias[n];
#pragma unroll
    for (int u = 0; u < 2; ++u)
#pragma unroll
      for (int r = 0; r < 8; ++r) {
        int m = m0 + u * 16 + rb + r;
        out[(size_t)m * D_DIM + n] = acc[u][t][r] + bv;
      }
  }
}

// ---------------------------------------------------------------------------
extern "C" void kernel_launch(void* const* d_in, const int* in_sizes, int n_in,
                              void* d_out, int out_size, void* d_ws,
                              size_t ws_size, hipStream_t stream) {
  const float* x    = (const float*)d_in[0];
  const float* Wqkv = (const float*)d_in[1];
  const float* bqkv = (const float*)d_in[2];
  const float* Wout = (const float*)d_in[3];
  const float* bout = (const float*)d_in[4];
  float* out = (float*)d_out;

  const int NX = B_SZ * S_LEN * D_DIM;  // 8388608
  const int NW1 = 3 * D_DIM * D_DIM;    // 3145728
  const int NW2 = D_DIM * D_DIM;        // 1048576

  char* ws = (char*)d_ws;
  _Float16* xh     = (_Float16*)ws;                                   // 16 MB
  _Float16* Wqkvh  = (_Float16*)(ws + (size_t)NX * 2);                //  6 MB
  _Float16* Wouth  = (_Float16*)(ws + (size_t)(NX + NW1) * 2);        //  2 MB
  char* ws2 = ws + (size_t)(NX + NW1 + NW2) * 2;
  const size_t SEG = (size_t)B_SZ * NH * S_LEN * DKH * sizeof(_Float16);  // 16MB
  _Float16* Qh    = (_Float16*)(ws2);
  _Float16* Kh    = (_Float16*)(ws2 + SEG);
  _Float16* Vt    = (_Float16*)(ws2 + 2 * SEG);
  _Float16* attnW = (_Float16*)(ws2 + 3 * SEG);

  cvt_kernel<<<NX / 4 / 256, 256, 0, stream>>>(x, xh, NX);
  cvt_kernel<<<NW1 / 4 / 256, 256, 0, stream>>>(Wqkv, Wqkvh, NW1);
  cvt_kernel<<<NW2 / 4 / 256, 256, 0, stream>>>(Wout, Wouth, NW2);

  dim3 blk(32);  // one wave32 per block
  qkv_kernel<<<dim3(256, 48), blk, 0, stream>>>(xh, Wqkvh, bqkv, Qh, Kh, Vt);
  attn_kernel<<<dim3(128, 64), blk, 0, stream>>>(Qh, Kh, Vt, attnW);
  proj_kernel<<<dim3(256, 16), blk, 0, stream>>>(attnW, Wouth, bout, out);
}
